// GPT_72739566125107
// MI455X (gfx1250) — compile-verified, matrix operand order checked by reading, not run on hardware
//
#include <hip/hip_runtime.h>

// ---------------- types & helpers ----------------
typedef __attribute__((ext_vector_type(16))) __bf16 bf16x16;
typedef __attribute__((ext_vector_type(8)))  float  f32x8;

__device__ __forceinline__ unsigned short f2bf(float f) {
  unsigned int u = __float_as_uint(f);
  u = (u + 0x7FFFu + ((u >> 16) & 1u)) >> 16;
  return (unsigned short)u;
}
__device__ __forceinline__ float bf2f(unsigned short s) {
  return __uint_as_float(((unsigned int)s) << 16);
}
__device__ __forceinline__ float gelu_f(float x) {
  return 0.5f * x * (1.0f + tanhf(0.7978845608028654f * (x + 0.044715f * x * x * x)));
}

// Load one 16x32 bf16 fragment (A layout; B uses the same layout on a
// [N][K] tile) from an LDS tile.  Matches the ISA striping:
//   lanes 0-15 : row = lane,     K pairs {0..7, 16..23}
//   lanes 16-31: row = lane-16,  K pairs {8..15, 24..31}
__device__ __forceinline__ bf16x16 load_frag(const unsigned short* base, int stride) {
  int lane = threadIdx.x & 31;
  const unsigned short* p = base + (lane & 15) * stride + (lane >> 4) * 8;
  union { unsigned int u[8]; bf16x16 v; } f;
#pragma unroll
  for (int j = 0; j < 8; ++j) {
    int k = (j < 4) ? (2 * j) : (16 + 2 * (j - 4));
    f.u[j] = *(const unsigned int*)(p + k);
  }
  return f.v;
}

__device__ __forceinline__ f32x8 wmma_bf16(bf16x16 a, bf16x16 b, f32x8 c) {
  return __builtin_amdgcn_wmma_f32_16x16x32_bf16(false, a, false, b, (short)0, c,
                                                 false, false);
}

// ---------------- model constants ----------------
#define Bc 8
#define Tc 64
#define Nc 16
#define DSc 128
#define Ac 32
#define Cc 1024
#define Hc 16
#define Lc 8
#define Sc 128          // 2*T
#define BNc 128         // B*N
#define Mtok 16384      // BN*S
#define Dh 64           // C/H

// ================= embed: tok + pos -> x (fp32) =================
__global__ __launch_bounds__(256) void embed_kernel(
    const float* __restrict__ states, const float* __restrict__ actions,
    const int* __restrict__ timesteps, const float* __restrict__ pos_emb,
    const float* __restrict__ gpe, const float* __restrict__ seW,
    const float* __restrict__ seb, const float* __restrict__ aeW,
    const float* __restrict__ aeb, float* __restrict__ x) {
  int row = blockIdx.x;            // bn*S + s
  int bn = row >> 7, s = row & 127;
  int b = bn >> 4, n = bn & 15, t = s >> 1;
  int tid = threadIdx.x;
  bool isState = (s & 1) == 0;
  int K = isState ? DSc : Ac;
  const float* src = isState
      ? (states  + (((size_t)b * Tc + t) * Nc + n) * DSc)
      : (actions + (((size_t)b * Tc + t) * Nc + n) * Ac);
  __shared__ float inbuf[DSc];
  if (tid < K) inbuf[tid] = src[tid];
  __syncthreads();
  int ts = timesteps[b * Tc + t];
  const float* Wm = isState ? seW : aeW;
  const float* bias = isState ? seb : aeb;
#pragma unroll
  for (int i = 0; i < 4; ++i) {
    int c = tid + i * 256;
    float acc = bias[c] + pos_emb[s * Cc + c] + gpe[(size_t)ts * Cc + c];
    for (int k = 0; k < K; ++k) acc += inbuf[k] * Wm[k * Cc + c];
    x[(size_t)row * Cc + c] = acc;
  }
}

// ================= layernorm: x(f32) -> h(bf16) =================
__global__ __launch_bounds__(256) void layernorm_kernel(
    const float* __restrict__ x, const float* __restrict__ w,
    const float* __restrict__ b, unsigned short* __restrict__ out) {
  int row = blockIdx.x, tid = threadIdx.x;
  const float* xr = x + (size_t)row * Cc;
  __shared__ float rs[256], rq[256];
  float vals[4], s = 0.f, s2 = 0.f;
#pragma unroll
  for (int i = 0; i < 4; ++i) {
    float v = xr[tid + i * 256];
    vals[i] = v; s += v; s2 += v * v;
  }
  rs[tid] = s; rq[tid] = s2; __syncthreads();
  for (int off = 128; off > 0; off >>= 1) {
    if (tid < off) { rs[tid] += rs[tid + off]; rq[tid] += rq[tid + off]; }
    __syncthreads();
  }
  float mean = rs[0] * (1.0f / Cc);
  float var = rq[0] * (1.0f / Cc) - mean * mean;
  float rstd = rsqrtf(var + 1e-5f);
#pragma unroll
  for (int i = 0; i < 4; ++i) {
    int c = tid + i * 256;
    out[(size_t)row * Cc + c] = f2bf((vals[i] - mean) * rstd * w[c] + b[c]);
  }
}

// ================= WMMA GEMM: Y = A(bf16)[M,K] @ W(f32->bf16)[K,N] + bias =================
// 8 waves, block tile 128x128, BK=32; wave tile 32x64 (2x4 WMMA accumulators).
// A tile staged via CDNA5 async global->LDS DMA (ASYNCcnt); B tile converted
// f32->bf16 in VALU while the DMA is in flight.
template <bool OUT_BF16, bool DO_GELU, bool DO_RES>
__global__ __launch_bounds__(256) void gemm_kernel(
    const unsigned short* __restrict__ A, const float* __restrict__ W,
    const float* __restrict__ bias, const float* __restrict__ residual,
    void* __restrict__ out, int M, int N, int K) {
  __shared__ unsigned short As[128][40];   // [m][k]  (row stride 80B = 5*16B)
  __shared__ unsigned short Bs[128][40];   // [n][k]  (B pre-transposed)
  int tid = threadIdx.x;
  int waveId = tid >> 5, lane = tid & 31;
  int waveM = waveId & 3, waveN = waveId >> 2;
  int blockRow = blockIdx.y * 128, blockCol = blockIdx.x * 128;

  f32x8 acc[2][4];
  const f32x8 zero = {0.f, 0.f, 0.f, 0.f, 0.f, 0.f, 0.f, 0.f};
#pragma unroll
  for (int i = 0; i < 2; ++i)
#pragma unroll
    for (int j = 0; j < 4; ++j) acc[i][j] = zero;

  int rA = tid >> 1, jbA = (tid & 1) * 16;              // 32B of the A tile / thread
  unsigned ldsA = (unsigned)(size_t)(void*)&As[rA][jbA]; // LDS byte offset (low 32b)

  for (int kb = 0; kb < K; kb += 32) {
    // ---- stage A tile: async global -> LDS (2 x b128 per thread) ----
    {
      const unsigned short* g = A + (size_t)(blockRow + rA) * K + kb + jbA;
#pragma unroll
      for (int i = 0; i < 2; ++i) {
        unsigned long long ga = (unsigned long long)(size_t)(g + i * 8);
        unsigned lo = ldsA + i * 16;
        asm volatile("global_load_async_to_lds_b128 %0, %1, off"
                     :: "v"(lo), "v"(ga) : "memory");
      }
    }
    // ---- stage B tile transposed, converting f32 -> bf16 on the fly ----
    {
      int n = tid & 127, kk0 = (tid >> 7) * 16;
#pragma unroll
      for (int i = 0; i < 16; ++i)
        Bs[n][kk0 + i] = f2bf(W[(size_t)(kb + kk0 + i) * N + blockCol + n]);
      if (kb + 32 < K)
        __builtin_prefetch(&W[(size_t)(kb + 32 + kk0) * N + blockCol + n], 0, 0);
    }
    asm volatile("s_wait_asynccnt 0" ::: "memory");   // this wave's DMAs landed
    __syncthreads();                                   // all waves' tiles visible

    bf16x16 afrag[2], bfrag[4];
#pragma unroll
    for (int tm = 0; tm < 2; ++tm)
      afrag[tm] = load_frag(&As[waveM * 32 + tm * 16][0], 40);
#pragma unroll
    for (int tn = 0; tn < 4; ++tn)
      bfrag[tn] = load_frag(&Bs[waveN * 64 + tn * 16][0], 40);
#pragma unroll
    for (int tm = 0; tm < 2; ++tm)
#pragma unroll
      for (int tn = 0; tn < 4; ++tn)
        acc[tm][tn] = wmma_bf16(afrag[tm], bfrag[tn], acc[tm][tn]);
    __syncthreads();
  }

  // epilogue (C/D layout: lanes 0-15 rows v, lanes 16-31 rows 8+v; col = lane%16)
  int half = lane >> 4, lcol = lane & 15;
#pragma unroll
  for (int tm = 0; tm < 2; ++tm)
#pragma unroll
    for (int tn = 0; tn < 4; ++tn) {
      int gm0 = blockRow + waveM * 32 + tm * 16 + half * 8;
      int gn = blockCol + waveN * 64 + tn * 16 + lcol;
      float bv = bias[gn];
#pragma unroll
      for (int v = 0; v < 8; ++v) {
        size_t idx = (size_t)(gm0 + v) * N + gn;
        float val = acc[tm][tn][v] + bv;
        if (DO_RES) val += residual[idx];
        if (DO_GELU) val = gelu_f(val);
        if (OUT_BF16) ((unsigned short*)out)[idx] = f2bf(val);
        else          ((float*)out)[idx] = val;
      }
    }
}

// ================= attention: per (bn, head), all-WMMA =================
__global__ __launch_bounds__(256) void attention_kernel(
    const unsigned short* __restrict__ q, const unsigned short* __restrict__ k,
    const unsigned short* __restrict__ v, unsigned short* __restrict__ out) {
  int bn = blockIdx.x >> 4, hd = blockIdx.x & 15;
  int tid = threadIdx.x, w = tid >> 5, lane = tid & 31;
  int half = lane >> 4, lcol = lane & 15;

  __shared__ union {
    struct { unsigned short qs[128][72]; unsigned short ks[128][72]; } a;
    unsigned short ps[128][136];
  } u;
  __shared__ unsigned short vt[64][136];   // V transposed: [dim][seq]

  // stage q,k ([seq][dim]) as dword pairs; v transposed elementwise
  {
    size_t base = ((size_t)bn * Sc) * Cc + hd * Dh;
#pragma unroll
    for (int i = 0; i < 16; ++i) {
      int idx = tid * 16 + i;                 // 4096 dword pairs
      int s = idx >> 5, cp = (idx & 31) * 2;
      *(unsigned int*)&u.a.qs[s][cp] =
          *(const unsigned int*)(q + base + (size_t)s * Cc + cp);
      *(unsigned int*)&u.a.ks[s][cp] =
          *(const unsigned int*)(k + base + (size_t)s * Cc + cp);
    }
#pragma unroll
    for (int i = 0; i < 32; ++i) {
      int idx = tid * 32 + i;
      int s = idx >> 6, c = idx & 63;
      vt[c][s] = v[base + (size_t)s * Cc + c];
    }
  }
  __syncthreads();

  // scores: wave w owns rows [w*16, w*16+16), all 128 cols; K=64 (2 steps)
  f32x8 sacc[8];
  const f32x8 zero = {0.f, 0.f, 0.f, 0.f, 0.f, 0.f, 0.f, 0.f};
#pragma unroll
  for (int tn = 0; tn < 8; ++tn) sacc[tn] = zero;
#pragma unroll
  for (int kk = 0; kk < 64; kk += 32) {
    bf16x16 af = load_frag(&u.a.qs[w * 16][kk], 72);
#pragma unroll
    for (int tn = 0; tn < 8; ++tn) {
      bf16x16 bf = load_frag(&u.a.ks[tn * 16][kk], 72);
      sacc[tn] = wmma_bf16(af, bf, sacc[tn]);
    }
  }
  __syncthreads();   // done reading qs/ks; ps may now alias them

  // in-register causal softmax (reduce over 16-lane halves via shfl_xor)
  float pr[8][8];    // [tn][vv]
#pragma unroll
  for (int vv = 0; vv < 8; ++vv) {
    int row = w * 16 + half * 8 + vv;
    float mx = -1e30f, sv[8];
#pragma unroll
    for (int tn = 0; tn < 8; ++tn) {
      float sc = sacc[tn][vv] * 0.125f;      // 1/sqrt(64)
      int col = tn * 16 + lcol;
      if (col > row) sc = -1e30f;
      sv[tn] = sc; mx = fmaxf(mx, sc);
    }
#pragma unroll
    for (int off = 1; off < 16; off <<= 1) mx = fmaxf(mx, __shfl_xor(mx, off, 32));
    float sum = 0.f;
#pragma unroll
    for (int tn = 0; tn < 8; ++tn) { sv[tn] = __expf(sv[tn] - mx); sum += sv[tn]; }
#pragma unroll
    for (int off = 1; off < 16; off <<= 1) sum += __shfl_xor(sum, off, 32);
    float inv = 1.0f / sum;
#pragma unroll
    for (int tn = 0; tn < 8; ++tn) pr[tn][vv] = sv[tn] * inv;
  }
#pragma unroll
  for (int vv = 0; vv < 8; ++vv) {
    int row = w * 16 + half * 8 + vv;
#pragma unroll
    for (int tn = 0; tn < 8; ++tn) u.ps[row][tn * 16 + lcol] = f2bf(pr[tn][vv]);
  }
  __syncthreads();

  // y = P @ V : wave w rows [w*16, +16), 64 cols; K=128 (4 steps)
  f32x8 yacc[4];
#pragma unroll
  for (int tn = 0; tn < 4; ++tn) yacc[tn] = zero;
#pragma unroll
  for (int kk = 0; kk < 128; kk += 32) {
    bf16x16 af = load_frag(&u.ps[w * 16][kk], 136);
#pragma unroll
    for (int tn = 0; tn < 4; ++tn) {
      bf16x16 bf = load_frag(&vt[tn * 16][kk], 136);
      yacc[tn] = wmma_bf16(af, bf, yacc[tn]);
    }
  }
  size_t obase = ((size_t)bn * Sc) * Cc + hd * Dh;
#pragma unroll
  for (int tn = 0; tn < 4; ++tn) {
    int col = tn * 16 + lcol;
#pragma unroll
    for (int vv = 0; vv < 8; ++vv) {
      int row = w * 16 + half * 8 + vv;
      out[obase + (size_t)row * Cc + col] = f2bf(yacc[tn][vv]);
    }
  }
}

// ================= final LN + head on row s = S-2 only =================
__global__ __launch_bounds__(256) void head_kernel(
    const float* __restrict__ x, const float* __restrict__ w,
    const float* __restrict__ b, const float* __restrict__ headW,
    float* __restrict__ out) {
  int bn = blockIdx.x, tid = threadIdx.x;
  const float* xr = x + ((size_t)bn * Sc + (Sc - 2)) * Cc;
  __shared__ float rs[256], rq[256], hn[Cc];
  float vals[4], s = 0.f, s2 = 0.f;
#pragma unroll
  for (int i = 0; i < 4; ++i) {
    float vv = xr[tid + i * 256];
    vals[i] = vv; s += vv; s2 += vv * vv;
  }
  rs[tid] = s; rq[tid] = s2; __syncthreads();
  for (int off = 128; off > 0; off >>= 1) {
    if (tid < off) { rs[tid] += rs[tid + off]; rq[tid] += rq[tid + off]; }
    __syncthreads();
  }
  float mean = rs[0] * (1.0f / Cc);
  float var = rq[0] * (1.0f / Cc) - mean * mean;
  float rstd = rsqrtf(var + 1e-5f);
#pragma unroll
  for (int i = 0; i < 4; ++i) {
    int c = tid + i * 256;
    hn[c] = (vals[i] - mean) * rstd * w[c] + b[c];
  }
  __syncthreads();
  if (tid < Ac) {
    float acc = 0.f;
    for (int c = 0; c < Cc; ++c) acc += hn[c] * headW[c * Ac + tid];
    out[bn * Ac + tid] = acc;
  }
}

// ================= host-side orchestration =================
extern "C" void kernel_launch(void* const* d_in, const int* in_sizes, int n_in,
                              void* d_out, int out_size, void* d_ws, size_t ws_size,
                              hipStream_t stream) {
  (void)in_sizes; (void)n_in; (void)out_size; (void)ws_size;
  const float* states  = (const float*)d_in[0];
  const float* actions = (const float*)d_in[1];
  const int*   tsteps  = (const int*)d_in[2];
  const float* pos_emb = (const float*)d_in[3];
  const float* gpe     = (const float*)d_in[4];
  const float* seW = (const float*)d_in[5];  const float* seb = (const float*)d_in[6];
  const float* aeW = (const float*)d_in[7];  const float* aeb = (const float*)d_in[8];
  const float* ln1w = (const float*)d_in[9];  const float* ln1b = (const float*)d_in[10];
  const float* Wq = (const float*)d_in[11];   const float* bq = (const float*)d_in[12];
  const float* Wk = (const float*)d_in[13];   const float* bk = (const float*)d_in[14];
  const float* Wv = (const float*)d_in[15];   const float* bv = (const float*)d_in[16];
  const float* Wp = (const float*)d_in[17];   const float* bp = (const float*)d_in[18];
  const float* ln2w = (const float*)d_in[19]; const float* ln2b = (const float*)d_in[20];
  const float* W1 = (const float*)d_in[21];   const float* b1 = (const float*)d_in[22];
  const float* W2 = (const float*)d_in[23];   const float* b2 = (const float*)d_in[24];
  const float* lnfw = (const float*)d_in[25]; const float* lnfb = (const float*)d_in[26];
  const float* headW = (const float*)d_in[27];

  char* ws = (char*)d_ws;
  float* x = (float*)ws;                    ws += (size_t)Mtok * Cc * 4;
  unsigned short* h   = (unsigned short*)ws; ws += (size_t)Mtok * Cc * 2;
  unsigned short* qb  = (unsigned short*)ws; ws += (size_t)Mtok * Cc * 2;
  unsigned short* kb_ = (unsigned short*)ws; ws += (size_t)Mtok * Cc * 2;
  unsigned short* vb  = (unsigned short*)ws; ws += (size_t)Mtok * Cc * 2;
  unsigned short* att = (unsigned short*)ws; ws += (size_t)Mtok * Cc * 2;
  unsigned short* mid = (unsigned short*)ws; ws += (size_t)Mtok * (4 * Cc) * 2;

  dim3 blk(256);
  embed_kernel<<<Mtok, blk, 0, stream>>>(states, actions, tsteps, pos_emb, gpe,
                                         seW, seb, aeW, aeb, x);

  dim3 gC(Cc / 128, Mtok / 128);        // N=1024
  dim3 g4C(4 * Cc / 128, Mtok / 128);   // N=4096

  for (int l = 0; l < Lc; ++l) {
    const float* wq = Wq + (size_t)l * Cc * Cc; const float* bq_ = bq + (size_t)l * Cc;
    const float* wk = Wk + (size_t)l * Cc * Cc; const float* bk2 = bk + (size_t)l * Cc;
    const float* wv = Wv + (size_t)l * Cc * Cc; const float* bv2 = bv + (size_t)l * Cc;
    const float* wp = Wp + (size_t)l * Cc * Cc; const float* bp2 = bp + (size_t)l * Cc;
    const float* w1 = W1 + (size_t)l * Cc * 4 * Cc; const float* b1_ = b1 + (size_t)l * 4 * Cc;
    const float* w2 = W2 + (size_t)l * 4 * Cc * Cc; const float* b2_ = b2 + (size_t)l * Cc;

    layernorm_kernel<<<Mtok, blk, 0, stream>>>(x, ln1w + (size_t)l * Cc,
                                               ln1b + (size_t)l * Cc, h);
    gemm_kernel<true,  false, false><<<gC, blk, 0, stream>>>(h, wq, bq_, nullptr, qb,  Mtok, Cc, Cc);
    gemm_kernel<true,  false, false><<<gC, blk, 0, stream>>>(h, wk, bk2, nullptr, kb_, Mtok, Cc, Cc);
    gemm_kernel<true,  false, false><<<gC, blk, 0, stream>>>(h, wv, bv2, nullptr, vb,  Mtok, Cc, Cc);
    attention_kernel<<<BNc * Hc, blk, 0, stream>>>(qb, kb_, vb, att);
    gemm_kernel<false, false, true ><<<gC, blk, 0, stream>>>(att, wp, bp2, x, x, Mtok, Cc, Cc);
    layernorm_kernel<<<Mtok, blk, 0, stream>>>(x, ln2w + (size_t)l * Cc,
                                               ln2b + (size_t)l * Cc, h);
    gemm_kernel<true,  true,  false><<<g4C, blk, 0, stream>>>(h, w1, b1_, nullptr, mid, Mtok, 4 * Cc, Cc);
    gemm_kernel<false, false, true ><<<gC, blk, 0, stream>>>(mid, w2, b2_, x, x, Mtok, Cc, 4 * Cc);
  }

  head_kernel<<<BNc, blk, 0, stream>>>(x, lnfw, lnfb, headW, (float*)d_out);
}
